// O3TensorProduct_18459769439023
// MI455X (gfx1250) — matrix operand order repaired and checked by reading, not compiled, not run on hardware
//
#include <hip/hip_runtime.h>

// O(3) tensor product (v==1): six 131072x128x128 GEMMs fused, bf16 WMMA with
// f32 accumulation on gfx1250 (CDNA5, wave32).
//
//   out0       = ALPHA*(y0*(x0@W1) + (INV_SQRT3*sum_i y1_i*x1_i)@W4) + b
//   out1[w][i] = ALPHA*(y1_i*(x0@W2)[w] + y0*(x1_i@W3)[w])
//
// Weights are pre-converted to bf16 and transposed (Wt[g][n][k] = W_g[k][n])
// so B-fragments for V_WMMA_F32_16X16X32_BF16 are contiguous 32B per lane and
// live persistently in VGPRs across the grid-stride tile loop.

#define MULW   128
#define PITCH  136    // bf16 elements per LDS panel row (128 + 8 pad -> no bank hotspot)

typedef __attribute__((ext_vector_type(16))) __bf16 v16bf;
typedef __attribute__((ext_vector_type(8)))  float  v8f;

union Frag  { int4 q[2]; v16bf v; };   // 32B: one 16-bf16 WMMA operand fragment
union Pack8 { __bf16 h[8]; int4 q; };  // 16B staging pack
union F8    { float4 v[2]; float f[8];  };
union F24   { float4 v[6]; float f[24]; };

__device__ __forceinline__ __bf16 f2bf(float f) {
  // Native conversion: clang emits v_cvt_pk_bf16_f32 when available on the
  // target, otherwise legalizes to the RNE shift sequence.
  return static_cast<__bf16>(f);
}

// ---------------------------------------------------------------------------
// Prep: W1..W4 (128x128 f32, [u][w]) -> bf16 transposed Wt[g][n][k] = W[k][n]
// ---------------------------------------------------------------------------
__global__ void o3tp_prep_weights(const float* __restrict__ W1,
                                  const float* __restrict__ W2,
                                  const float* __restrict__ W3,
                                  const float* __restrict__ W4,
                                  __bf16* __restrict__ wt) {
  int id  = blockIdx.x * blockDim.x + threadIdx.x;  // 0 .. 65535
  int g   = id >> 14;
  int rem = id & 16383;
  int n   = rem >> 7;
  int k   = rem & 127;
  const float* W = (g == 0) ? W1 : (g == 1) ? W2 : (g == 2) ? W3 : W4;
  wt[id] = f2bf(W[k * MULW + n]);
}

// ---------------------------------------------------------------------------
// Main kernel: 256 threads = 8 waves; wave w owns output columns [16w,16w+16).
// B fragments (4 weights x 4 K-steps) persist in VGPRs across the tile loop.
// ---------------------------------------------------------------------------
__global__ __launch_bounds__(256) void o3tp_main(
    const float*  __restrict__ x,
    const float*  __restrict__ y,
    const __bf16* __restrict__ wt,
    const float*  __restrict__ b,
    float*        __restrict__ out,
    int ntiles)
{
  __shared__ __align__(16) __bf16 panels[5][16][PITCH];  // x0, x1_0..2, S  (~21.3 KB)
  __shared__ float ly[16][4];                            // per-node y0,y1x,y1y,y1z

  const int tid  = threadIdx.x;
  const int lane = tid & 31;
  const int wave = tid >> 5;
  const int col  = lane & 15;       // also A-matrix row M for this lane
  const int hi   = lane >> 4;       // lane half selects K sub-chunk / D row half
  const int wch  = (wave << 4) + col;  // output channel 0..127 (B column N)

  // ---- persistent B fragments: Bf[g][ks], lane holds column wch, 16 K-values
  Frag Bf[4][4];
  #pragma unroll
  for (int g = 0; g < 4; ++g) {
    #pragma unroll
    for (int ks = 0; ks < 4; ++ks) {
      const __bf16* p = wt + ((g << 14) | (wch << 7) | (ks << 5) | (hi << 4));
      Bf[g][ks].q[0] = *(const int4*)(p);
      Bf[g][ks].q[1] = *(const int4*)(p + 8);
    }
  }
  const float bias = b[wch];

  // staging mapping: thread -> (node row sm, u-chunk of 8)
  const int sm = tid >> 4;
  const int sc = tid & 15;
  const int u0 = sc << 3;

  const float INV_SQRT3 = 0.57735026918962576f;
  const float ALPHA     = 0.0625f;   // 1/sqrt(2*128)

  for (int tile = blockIdx.x; tile < ntiles; tile += gridDim.x) {
    const int node0 = tile << 4;

    // Prefetch next grid-stride tile's x rows into cache (global_prefetch_b8)
    // so the next staging phase hits near caches while this tile computes.
    const int ntile = tile + gridDim.x;
    if (ntile < ntiles) {
      const float* xn = x + (size_t)((ntile << 4) + sm) * (4 * MULW);
      __builtin_prefetch(xn + u0, 0, 1);
      __builtin_prefetch(xn + MULW + 3 * u0, 0, 1);
    }

    // ---------------- stage A panels: fp32 -> bf16 into LDS ----------------
    {
      const float* xr = x + (size_t)(node0 + sm) * (4 * MULW);
      const float* yr = y + (size_t)(node0 + sm) * 4;
      const float y1a = yr[1], y1b = yr[2], y1c = yr[3];
      if (sc == 0) {
        ly[sm][0] = yr[0]; ly[sm][1] = y1a; ly[sm][2] = y1b; ly[sm][3] = y1c;
      }
      F8 X0;
      X0.v[0] = *(const float4*)(xr + u0);
      X0.v[1] = *(const float4*)(xr + u0 + 4);
      F24 X1;
      const float* x1p = xr + MULW + 3 * u0;
      #pragma unroll
      for (int j = 0; j < 6; ++j) X1.v[j] = *(const float4*)(x1p + 4 * j);

      Pack8 p0, p1, p2, p3, ps;
      #pragma unroll
      for (int u = 0; u < 8; ++u) {
        const float xa = X1.f[3 * u + 0];
        const float xb = X1.f[3 * u + 1];
        const float xc = X1.f[3 * u + 2];
        p0.h[u] = f2bf(X0.f[u]);
        p1.h[u] = f2bf(xa);
        p2.h[u] = f2bf(xb);
        p3.h[u] = f2bf(xc);
        ps.h[u] = f2bf(INV_SQRT3 * (y1a * xa + y1b * xb + y1c * xc));
      }
      *(int4*)&panels[0][sm][u0] = p0.q;
      *(int4*)&panels[1][sm][u0] = p1.q;
      *(int4*)&panels[2][sm][u0] = p2.q;
      *(int4*)&panels[3][sm][u0] = p3.q;
      *(int4*)&panels[4][sm][u0] = ps.q;
    }
    __syncthreads();

    // ---------------- compute: 4 K-steps x 6 WMMAs ----------------
    v8f c1 = {0,0,0,0,0,0,0,0}, c2 = {0,0,0,0,0,0,0,0};
    v8f d0 = {0,0,0,0,0,0,0,0}, d1 = {0,0,0,0,0,0,0,0};
    v8f d2 = {0,0,0,0,0,0,0,0}, s4 = {0,0,0,0,0,0,0,0};
    #pragma unroll
    for (int ks = 0; ks < 4; ++ks) {
      const int k0 = (ks << 5) + (hi << 4);   // A layout: lane<16 -> K[0..15], lane>=16 -> K[16..31]
      Frag A0, A1, A2, A3, AS;
      A0.q[0] = *(const int4*)&panels[0][col][k0];
      A0.q[1] = *(const int4*)&panels[0][col][k0 + 8];
      A1.q[0] = *(const int4*)&panels[1][col][k0];
      A1.q[1] = *(const int4*)&panels[1][col][k0 + 8];
      A2.q[0] = *(const int4*)&panels[2][col][k0];
      A2.q[1] = *(const int4*)&panels[2][col][k0 + 8];
      A3.q[0] = *(const int4*)&panels[3][col][k0];
      A3.q[1] = *(const int4*)&panels[3][col][k0 + 8];
      AS.q[0] = *(const int4*)&panels[4][col][k0];
      AS.q[1] = *(const int4*)&panels[4][col][k0 + 8];

      c1 = __builtin_amdgcn_wmma_f32_16x16x32_bf16(false, A0.v, false, Bf[0][ks].v, (short)0, c1, false, false);
      c2 = __builtin_amdgcn_wmma_f32_16x16x32_bf16(false, A0.v, false, Bf[1][ks].v, (short)0, c2, false, false);
      d0 = __builtin_amdgcn_wmma_f32_16x16x32_bf16(false, A1.v, false, Bf[2][ks].v, (short)0, d0, false, false);
      d1 = __builtin_amdgcn_wmma_f32_16x16x32_bf16(false, A2.v, false, Bf[2][ks].v, (short)0, d1, false, false);
      d2 = __builtin_amdgcn_wmma_f32_16x16x32_bf16(false, A3.v, false, Bf[2][ks].v, (short)0, d2, false, false);
      s4 = __builtin_amdgcn_wmma_f32_16x16x32_bf16(false, AS.v, false, Bf[3][ks].v, (short)0, s4, false, false);
    }

    // ---------------- combine scalars + store ----------------
    // D layout: VGPR r, lanes 0-15 -> row r, lanes 16-31 -> row 8+r; column = col.
    #pragma unroll
    for (int r = 0; r < 8; ++r) {
      const int   mm  = (hi << 3) + r;
      const float y0v = ly[mm][0];
      const float ya  = ly[mm][1];
      const float yb  = ly[mm][2];
      const float yc  = ly[mm][3];
      float* orow = out + (size_t)(node0 + mm) * (4 * MULW);
      orow[wch] = ALPHA * (y0v * c1[r] + s4[r]) + bias;
      float* o1 = orow + MULW + 3 * wch;
      o1[0] = ALPHA * (ya * c2[r] + y0v * d0[r]);
      o1[1] = ALPHA * (yb * c2[r] + y0v * d1[r]);
      o1[2] = ALPHA * (yc * c2[r] + y0v * d2[r]);
    }
    __syncthreads();   // protect panels before next tile's staging
  }
}

// ---------------------------------------------------------------------------
extern "C" void kernel_launch(void* const* d_in, const int* in_sizes, int n_in,
                              void* d_out, int out_size, void* d_ws, size_t ws_size,
                              hipStream_t stream) {
  const float* x  = (const float*)d_in[0];
  const float* y  = (const float*)d_in[1];
  const float* W1 = (const float*)d_in[2];
  const float* W2 = (const float*)d_in[3];
  const float* W3 = (const float*)d_in[4];
  const float* W4 = (const float*)d_in[5];
  const float* b  = (const float*)d_in[6];
  float*  out = (float*)d_out;
  __bf16* wt  = (__bf16*)d_ws;          // needs 4*128*128*2 = 128 KB scratch

  const int n      = in_sizes[0] / (4 * MULW);
  const int ntiles = n / 16;

  o3tp_prep_weights<<<256, 256, 0, stream>>>(W1, W2, W3, W4, wt);

  int blocks = ntiles < 2048 ? ntiles : 2048;
  o3tp_main<<<blocks, 256, 0, stream>>>(x, y, wt, b, out, ntiles);
}
